// HilbertReNetLayer_80762565034210
// MI455X (gfx1250) — compile-verified, matrix operand order checked by reading, not run on hardware
//
#include <hip/hip_runtime.h>
#include <hip/hip_fp16.h>

// ---------------------------------------------------------------------------
// HilbertReNet on gfx1250 (MI455X), wave32 + v_wmma_f32_16x16x32_f16.
//   1) Hilbert board (reference recursion) + inverse permutation
//   2) pack patches -> X1[t][n][12] f16 (replicates reference view math)
//   3) weights -> f16 (layer-1 Wih padded K=12->32), biases fused
//   4) gru_layer1: 512 waves = (16-row tile, direction); X staged into LDS
//      with GLOBAL_LOAD_ASYNC_TO_LDS_B32 (ASYNCcnt), WMMA input+hidden GEMM,
//      trans-op gates, relu output scattered into layer-2 layout (f16)
//   5) gru_layer2: F=128 input GEMM A-frags straight from global; fp32 out
// ---------------------------------------------------------------------------

#define BATCH 32
#define CH    3
#define IMG   256
#define PP    128      // patch grid P
#define F1    12
#define F1P   32       // padded K for layer-1 input GEMM
#define HID   64
#define GG    192      // 3*HID gate columns
#define TT    128      // sequence length
#define NSEQ  4096     // P*B sequences per layer
#define F2    128      // 2*HID

typedef __attribute__((ext_vector_type(16))) _Float16 v16h;
typedef __attribute__((ext_vector_type(8)))  _Float16 v8h;
typedef __attribute__((ext_vector_type(8)))  float    v8f;

union HF16 { v16h v; v8h h[2]; };

// Gather a 16x32 f16 WMMA A/B fragment from a row-major matrix.
// Lane = hi*16 + m16 holds row `row`; elements 0..7 -> K = kbase + hi*8,
// elements 8..15 -> K = kbase + 16 + hi*8 (ISA 7.12.2).
__device__ __forceinline__ v16h load_frag(const _Float16* base, int row,
                                          int rstride, int kbase, int hi) {
  const _Float16* p = base + (size_t)row * rstride + kbase + hi * 8;
  HF16 u;
  u.h[0] = *(const v8h*)p;
  u.h[1] = *(const v8h*)(p + 16);
  return u.v;
}

#define WMMA(A, B, C) \
  __builtin_amdgcn_wmma_f32_16x16x32_f16(false, (A), false, (B), (short)0, (C), false, false)

// Fast gate math on the trans pipe (co-executes with XDL WMMA).
__device__ __forceinline__ float fexp2(float x) { return __builtin_amdgcn_exp2f(x); }
__device__ __forceinline__ float frcp(float x)  { return __builtin_amdgcn_rcpf(x); }
__device__ __forceinline__ float sigm(float x) {
  return frcp(1.0f + fexp2(x * -1.4426950408889634f));
}
__device__ __forceinline__ float ftanh(float x) {
#if __has_builtin(__builtin_amdgcn_tanhf)
  return __builtin_amdgcn_tanhf(x);
#elif __has_builtin(__builtin_amdgcn_tanh_f32)
  return __builtin_amdgcn_tanh_f32(x);
#else
  float e = fexp2(x * 2.8853900817779268f);   // e^(2x)
  return (e - 1.0f) * frcp(e + 1.0f);
#endif
}

// Generic (__shared__-derived) pointer -> LDS byte offset: low 32 bits of the
// shared-aperture flat address are the in-wave LDS offset.
__device__ __forceinline__ unsigned lds_off(const void* p) {
  return (unsigned)(unsigned long long)p;
}

// ---------------------------------------------------------------------------
// Hilbert curve construction (replays the reference recursion exactly).
// ---------------------------------------------------------------------------
__global__ void k_hilbert_init(int* b) {
  if (blockIdx.x == 0 && threadIdx.x == 0) { b[0] = 0; b[1] = 3; b[2] = 1; b[3] = 2; }
}

__global__ void k_hilbert_step(const int* __restrict__ src, int* __restrict__ dst, int s) {
  int n = 2 * s;
  int tid = blockIdx.x * blockDim.x + threadIdx.x;
  if (tid >= n * n) return;
  int r = tid / n, c = tid % n;
  int s2 = s * s;
  int v;
  if (r < s && c < s)            v = src[c * s + r];                                    // hp.T
  else if (r >= s && c < s)      v = src[(r - s) * s + c] + s2;                         // hp+n
  else if (r >= s && c >= s)     v = src[(r - s) * s + (c - s)] + 2 * s2;               // hp+2n
  else                           v = src[(s - 1 - (c - s)) * s + (s - 1 - r)] + 3 * s2; // rot180(hp+3n).T
  dst[r * n + c] = v;
}

__global__ void k_invperm(const int* __restrict__ board, int* __restrict__ idx) {
  int p = blockIdx.x * blockDim.x + threadIdx.x;
  if (p < PP * PP) idx[board[p]] = p;   // idx = argsort(board.flat)
}

// ---------------------------------------------------------------------------
// Pack: x (B,C,256,256) -> X1[t=j][n=i*32+b][f] f16, with the reference's
// row-major re-view of (C,P,P,2,2) as (12,P,P) and the Hilbert gather.
// ---------------------------------------------------------------------------
__global__ void k_pack(const float* __restrict__ x, const int* __restrict__ idx,
                       _Float16* __restrict__ X1) {
  int tid = blockIdx.x * blockDim.x + threadIdx.x;
  const int total = TT * NSEQ * F1;
  if (tid >= total) return;
  int f = tid % F1;
  int n = (tid / F1) & (NSEQ - 1);
  int t = tid / (F1 * NSEQ);
  int i = n >> 5, b = n & 31, j = t;
  int q = idx[i * PP + j];               // new grid (i,j) <- old flat q
  int i0 = q >> 7, j0 = q & 127;
  int g = f * (PP * PP) + i0 * PP + j0;  // flat into (12,P,P) view
  int ch  = g >> 16;                     // / (4*P*P)
  int rem = g & 65535;
  int pq = rem >> 2, s = rem & 3;
  int r = pq >> 7, c = pq & 127;
  int prow = r * 2 + (s >> 1), pcol = c * 2 + (s & 1);
  float v = x[(((size_t)b * CH + ch) * IMG + prow) * IMG + pcol];
  X1[tid] = (_Float16)v;
}

// ---------------------------------------------------------------------------
// Weight conversion fp32 -> f16 with K padding; bias fusion.
// ---------------------------------------------------------------------------
__global__ void k_w2h(const float* __restrict__ src, _Float16* __restrict__ dst,
                      int rows, int K, int Kp) {
  int t = blockIdx.x * blockDim.x + threadIdx.x;
  if (t >= rows * Kp) return;
  int r = t / Kp, c = t % Kp;
  dst[t] = (_Float16)((c < K) ? src[r * K + c] : 0.0f);
}

// BA[g<128] = bih+bhh (r,z fused); BA[g>=128] = bih (xn); BH = bhh_n.
__global__ void k_bias(const float* __restrict__ bih, const float* __restrict__ bhh,
                       float* __restrict__ BA, float* __restrict__ BH) {
  int g = blockIdx.x * blockDim.x + threadIdx.x;
  if (g < GG)  BA[g] = (g < 2 * HID) ? (bih[g] + bhh[g]) : bih[g];
  if (g < HID) BH[g] = bhh[2 * HID + g];
}

// ---------------------------------------------------------------------------
// GRU layer 1 (vertical scan). 512 waves: wid = tile*2 + dir.
// ---------------------------------------------------------------------------
__global__ void __launch_bounds__(128) gru_layer1(
    const _Float16* __restrict__ X1, _Float16* __restrict__ X2,
    const _Float16* __restrict__ Wih, const _Float16* __restrict__ Whh,
    const float* __restrict__ BA, const float* __restrict__ BH) {
  __shared__ __align__(32) _Float16 sh_h[4][16 * HID];
  __shared__ __align__(32) _Float16 sh_x[4][16 * F1P];
  const int wave = threadIdx.x >> 5;
  const int lane = threadIdx.x & 31;
  const int m16 = lane & 15;
  const int hi  = lane >> 4;
  const int wid  = blockIdx.x * 4 + wave;   // 0..511
  const int tile = wid >> 1;                // 0..255
  const int dir  = wid & 1;
  const int n0   = tile * 16;
  const int irow = n0 >> 5;                 // patch row i (constant per tile)
  const int b0   = n0 & 31;
  _Float16* hbuf = sh_h[wave];
  _Float16* xbuf = sh_x[wave];
  const unsigned xoff = lds_off(xbuf);

  // zero the K-pad columns of the staged X tile once
  for (int q = lane; q < 16 * (F1P - F1); q += 32) {
    int r = q / (F1P - F1), c = q % (F1P - F1);
    xbuf[r * F1P + F1 + c] = (_Float16)0.0f;
  }

  const _Float16* wih = Wih + (size_t)dir * GG * F1P;
  const _Float16* whh = Whh + (size_t)dir * GG * HID;
  const float* ba = BA + dir * GG;
  const float* bh = BH + dir * HID;
  float biasA[12], biasH[4];
#pragma unroll
  for (int nt = 0; nt < 12; ++nt) biasA[nt] = ba[nt * 16 + m16];
#pragma unroll
  for (int j = 0; j < 4; ++j) biasH[j] = bh[j * 16 + m16];

  v8f hreg[4];
#pragma unroll
  for (int j = 0; j < 4; ++j)
#pragma unroll
    for (int v = 0; v < 8; ++v) {
      hreg[j][v] = 0.0f;
      hbuf[(v + 8 * hi) * HID + j * 16 + m16] = (_Float16)0.0f;
    }

  for (int step = 0; step < TT; ++step) {
    const int t = dir ? (TT - 1 - step) : step;

    // Async-stage the 16x12 X tile into LDS: 96 dwords, 3 wave-wide issues.
    // GV mode: VDST = per-lane LDS byte offset, VADDR = 64-bit global addr.
#pragma unroll
    for (int it = 0; it < 3; ++it) {
      int q = it * 32 + lane;            // 0..95
      int row = q / 6, d = q % 6;        // 6 dwords per 24B row
      unsigned ldsa = xoff + row * (F1P * 2) + d * 4;
      unsigned long long ga =
          (unsigned long long)(const void*)(X1 + ((size_t)t * NSEQ + n0 + row) * F1) +
          (unsigned)(d * 4);
      asm volatile("global_load_async_to_lds_b32 %0, %1, off"
                   :: "v"(ldsa), "v"(ga) : "memory");
    }
    // prefetch next timestep's rows
    {
      int tn = dir ? (t - 1) : (t + 1);
      if ((unsigned)tn < TT)
        __builtin_prefetch(X1 + ((size_t)tn * NSEQ + n0 + m16) * F1, 0, 1);
    }

    v8f acc[12], accH[4];
#pragma unroll
    for (int nt = 0; nt < 12; ++nt)
#pragma unroll
      for (int v = 0; v < 8; ++v) acc[nt][v] = biasA[nt];
#pragma unroll
    for (int j = 0; j < 4; ++j)
#pragma unroll
      for (int v = 0; v < 8; ++v) accH[j][v] = biasH[j];

    // wait for the async LDS fill before the fragment gather
    asm volatile("s_wait_asynccnt 0x0" ::: "memory");

    // input GEMM (K=32, padded)
    {
      v16h xf = load_frag(xbuf, m16, F1P, 0, hi);
#pragma unroll
      for (int nt = 0; nt < 12; ++nt) {
        v16h bf = load_frag(wih, nt * 16 + m16, F1P, 0, hi);
        acc[nt] = WMMA(xf, bf, acc[nt]);
      }
    }
    // hidden GEMM (K=64): r,z into acc[0..7]; n-gate h-part into accH
#pragma unroll
    for (int kb = 0; kb < 2; ++kb) {
      v16h hf = load_frag(hbuf, m16, HID, kb * 32, hi);
#pragma unroll
      for (int nt = 0; nt < 8; ++nt) {
        v16h bf = load_frag(whh, nt * 16 + m16, HID, kb * 32, hi);
        acc[nt] = WMMA(hf, bf, acc[nt]);
      }
#pragma unroll
      for (int j = 0; j < 4; ++j) {
        v16h bf = load_frag(whh, (8 + j) * 16 + m16, HID, kb * 32, hi);
        accH[j] = WMMA(hf, bf, accH[j]);
      }
    }

    // gates + state update + scatter relu(h) into layer-2 layout
#pragma unroll
    for (int j = 0; j < 4; ++j) {
#pragma unroll
      for (int v = 0; v < 8; ++v) {
        float r = sigm(acc[j][v]);
        float z = sigm(acc[4 + j][v]);
        float nn = ftanh(acc[8 + j][v] + r * accH[j][v]);
        float h = (1.0f - z) * nn + z * hreg[j][v];
        hreg[j][v] = h;
        int m = v + 8 * hi;
        hbuf[m * HID + j * 16 + m16] = (_Float16)h;
        float o = h > 0.0f ? h : 0.0f;
        int n2 = t * BATCH + (b0 + m);          // layer-2 sequence index
        int f = dir * HID + j * 16 + m16;
        X2[((size_t)irow * NSEQ + n2) * F2 + f] = (_Float16)o;
      }
    }
  }
}

// ---------------------------------------------------------------------------
// GRU layer 2 (horizontal scan). F=128 input GEMM, A-frags from global.
// ---------------------------------------------------------------------------
__global__ void __launch_bounds__(128) gru_layer2(
    const _Float16* __restrict__ X2, float* __restrict__ OUT,
    const _Float16* __restrict__ Wih, const _Float16* __restrict__ Whh,
    const float* __restrict__ BA, const float* __restrict__ BH) {
  __shared__ __align__(32) _Float16 sh_h[4][16 * HID];
  const int wave = threadIdx.x >> 5;
  const int lane = threadIdx.x & 31;
  const int m16 = lane & 15;
  const int hi  = lane >> 4;
  const int wid  = blockIdx.x * 4 + wave;
  const int tile = wid >> 1;
  const int dir  = wid & 1;
  const int n0   = tile * 16;
  const int jcol = n0 >> 5;                 // patch col j (constant per tile)
  const int b0   = n0 & 31;
  _Float16* hbuf = sh_h[wave];

  const _Float16* wih = Wih + (size_t)dir * GG * F2;
  const _Float16* whh = Whh + (size_t)dir * GG * HID;
  const float* ba = BA + dir * GG;
  const float* bh = BH + dir * HID;
  float biasA[12], biasH[4];
#pragma unroll
  for (int nt = 0; nt < 12; ++nt) biasA[nt] = ba[nt * 16 + m16];
#pragma unroll
  for (int j = 0; j < 4; ++j) biasH[j] = bh[j * 16 + m16];

  v8f hreg[4];
#pragma unroll
  for (int j = 0; j < 4; ++j)
#pragma unroll
    for (int v = 0; v < 8; ++v) {
      hreg[j][v] = 0.0f;
      hbuf[(v + 8 * hi) * HID + j * 16 + m16] = (_Float16)0.0f;
    }

  for (int step = 0; step < TT; ++step) {
    const int t = dir ? (TT - 1 - step) : step;
    const _Float16* xbase = X2 + ((size_t)t * NSEQ + n0) * F2;
    {
      int tn = dir ? (t - 1) : (t + 1);
      if ((unsigned)tn < TT)
        __builtin_prefetch(X2 + ((size_t)tn * NSEQ + n0 + m16) * F2 + hi * 64, 0, 1);
    }

    v8f acc[12], accH[4];
#pragma unroll
    for (int nt = 0; nt < 12; ++nt)
#pragma unroll
      for (int v = 0; v < 8; ++v) acc[nt][v] = biasA[nt];
#pragma unroll
    for (int j = 0; j < 4; ++j)
#pragma unroll
      for (int v = 0; v < 8; ++v) accH[j][v] = biasH[j];

    // input GEMM: K=128 = 4 WMMA K-blocks
#pragma unroll
    for (int kb = 0; kb < 4; ++kb) {
      v16h xf = load_frag(xbase, m16, F2, kb * 32, hi);
#pragma unroll
      for (int nt = 0; nt < 12; ++nt) {
        v16h bf = load_frag(wih, nt * 16 + m16, F2, kb * 32, hi);
        acc[nt] = WMMA(xf, bf, acc[nt]);
      }
    }
    // hidden GEMM
#pragma unroll
    for (int kb = 0; kb < 2; ++kb) {
      v16h hf = load_frag(hbuf, m16, HID, kb * 32, hi);
#pragma unroll
      for (int nt = 0; nt < 8; ++nt) {
        v16h bf = load_frag(whh, nt * 16 + m16, HID, kb * 32, hi);
        acc[nt] = WMMA(hf, bf, acc[nt]);
      }
#pragma unroll
      for (int j = 0; j < 4; ++j) {
        v16h bf = load_frag(whh, (8 + j) * 16 + m16, HID, kb * 32, hi);
        accH[j] = WMMA(hf, bf, accH[j]);
      }
    }

#pragma unroll
    for (int j = 0; j < 4; ++j) {
#pragma unroll
      for (int v = 0; v < 8; ++v) {
        float r = sigm(acc[j][v]);
        float z = sigm(acc[4 + j][v]);
        float nn = ftanh(acc[8 + j][v] + r * accH[j][v]);
        float h = (1.0f - z) * nn + z * hreg[j][v];
        hreg[j][v] = h;
        int m = v + 8 * hi;
        hbuf[m * HID + j * 16 + m16] = (_Float16)h;
        int b = b0 + m;
        int f = dir * HID + j * 16 + m16;
        // out[b][f][i=t][j=jcol]
        OUT[(((size_t)b * F2 + f) * PP + t) * PP + jcol] = h;
      }
    }
  }
}

// ---------------------------------------------------------------------------
extern "C" void kernel_launch(void* const* d_in, const int* in_sizes, int n_in,
                              void* d_out, int out_size, void* d_ws, size_t ws_size,
                              hipStream_t stream) {
  const float* x     = (const float*)d_in[0];
  const float* w1ihf = (const float*)d_in[1];
  const float* w1hhf = (const float*)d_in[2];
  const float* b1ihf = (const float*)d_in[3];
  const float* b1hhf = (const float*)d_in[4];
  const float* w1ihb = (const float*)d_in[5];
  const float* w1hhb = (const float*)d_in[6];
  const float* b1ihb = (const float*)d_in[7];
  const float* b1hhb = (const float*)d_in[8];
  const float* w2ihf = (const float*)d_in[9];
  const float* w2hhf = (const float*)d_in[10];
  const float* b2ihf = (const float*)d_in[11];
  const float* b2hhf = (const float*)d_in[12];
  const float* w2ihb = (const float*)d_in[13];
  const float* w2hhb = (const float*)d_in[14];
  const float* b2ihb = (const float*)d_in[15];
  const float* b2hhb = (const float*)d_in[16];
  (void)in_sizes; (void)n_in; (void)out_size; (void)ws_size;

  char* ws = (char*)d_ws;
  size_t off = 0;
  auto take = [&](size_t bytes) -> char* {
    char* p = ws + off;
    off = (off + bytes + 255) & ~(size_t)255;
    return p;
  };

  _Float16* X1   = (_Float16*)take((size_t)TT * NSEQ * F1 * 2);   // 12.6 MB
  _Float16* X2   = (_Float16*)take((size_t)TT * NSEQ * F2 * 2);   // 128 MB
  int* bA        = (int*)take(PP * PP * 4);
  int* bB        = (int*)take(PP * PP * 4);
  int* idx       = (int*)take(PP * PP * 4);
  _Float16* W1ih = (_Float16*)take((size_t)2 * GG * F1P * 2);
  _Float16* W1hh = (_Float16*)take((size_t)2 * GG * HID * 2);
  _Float16* W2ih = (_Float16*)take((size_t)2 * GG * F2 * 2);
  _Float16* W2hh = (_Float16*)take((size_t)2 * GG * HID * 2);
  float* BA1     = (float*)take(2 * GG * 4);
  float* BH1     = (float*)take(2 * HID * 4);
  float* BA2     = (float*)take(2 * GG * 4);
  float* BH2     = (float*)take(2 * HID * 4);

  // Hilbert board (2x2 -> 128x128), then inverse permutation
  k_hilbert_init<<<1, 32, 0, stream>>>(bA);
  int* bufs[2] = {bA, bB};
  int s = 2;
  for (int k = 0; k < 6; ++k) {
    int n = 2 * s, tot = n * n;
    k_hilbert_step<<<(tot + 255) / 256, 256, 0, stream>>>(bufs[k & 1], bufs[(k + 1) & 1], s);
    s = n;
  }
  // 6 steps: A->B->A->B->A->B->A, final board in bA
  k_invperm<<<(PP * PP + 255) / 256, 256, 0, stream>>>(bA, idx);

  {
    int tot = TT * NSEQ * F1;
    k_pack<<<(tot + 255) / 256, 256, 0, stream>>>(x, idx, X1);
  }

  // weights -> f16 (layer-1 Wih padded to K=32)
  k_w2h<<<(GG * F1P + 255) / 256, 256, 0, stream>>>(w1ihf, W1ih,            GG, F1,  F1P);
  k_w2h<<<(GG * F1P + 255) / 256, 256, 0, stream>>>(w1ihb, W1ih + GG * F1P, GG, F1,  F1P);
  k_w2h<<<(GG * HID + 255) / 256, 256, 0, stream>>>(w1hhf, W1hh,            GG, HID, HID);
  k_w2h<<<(GG * HID + 255) / 256, 256, 0, stream>>>(w1hhb, W1hh + GG * HID, GG, HID, HID);
  k_w2h<<<(GG * F2  + 255) / 256, 256, 0, stream>>>(w2ihf, W2ih,            GG, F2,  F2);
  k_w2h<<<(GG * F2  + 255) / 256, 256, 0, stream>>>(w2ihb, W2ih + GG * F2,  GG, F2,  F2);
  k_w2h<<<(GG * HID + 255) / 256, 256, 0, stream>>>(w2hhf, W2hh,            GG, HID, HID);
  k_w2h<<<(GG * HID + 255) / 256, 256, 0, stream>>>(w2hhb, W2hh + GG * HID, GG, HID, HID);
  k_bias<<<1, 256, 0, stream>>>(b1ihf, b1hhf, BA1,      BH1);
  k_bias<<<1, 256, 0, stream>>>(b1ihb, b1hhb, BA1 + GG, BH1 + HID);
  k_bias<<<1, 256, 0, stream>>>(b2ihf, b2hhf, BA2,      BH2);
  k_bias<<<1, 256, 0, stream>>>(b2ihb, b2hhb, BA2 + GG, BH2 + HID);

  // 512 (tile, dir) waves per layer => 128 blocks of 4 waves
  gru_layer1<<<128, 128, 0, stream>>>(X1, X2, W1ih, W1hh, BA1, BH1);
  gru_layer2<<<128, 128, 0, stream>>>(X2, (float*)d_out, W2ih, W2hh, BA2, BH2);
}
